// Cross_Attention_32427003085113
// MI455X (gfx1250) — matrix operand order repaired and checked by reading, not compiled, not run on hardware
//
#include <hip/hip_runtime.h>
#include <hip/hip_bf16.h>

typedef __bf16 bf16_t;
typedef __attribute__((ext_vector_type(16))) __bf16 v16bf;
typedef __attribute__((ext_vector_type(8)))  __bf16 v8bf;
typedef __attribute__((ext_vector_type(8)))  float  v8f;

#define WMMA_BF16(a,b,c) __builtin_amdgcn_wmma_f32_16x16x32_bf16(false,(a),false,(b),(short)0,(c),false,false)

constexpr int Bb    = 2;
constexpr int Nn    = 2048;
constexpr int Mm    = 2048;
constexpr int DIM   = 512;
constexpr int H     = 8;
constexpr int DH    = 64;
constexpr int INNER = 512;

// ---------------- fragment loaders (ISA 7.12.2 layouts) ----------------
// A-matrix 16x32 bf16, row-major source, leading-dim ld (elements).
__device__ __forceinline__ v16bf load_a_frag(const bf16_t* base, int ld, int kb, int lane) {
    int r = lane & 15, h = lane >> 4;
    const bf16_t* p = base + (size_t)r * ld + kb + h * 8;
    v8bf lo = *(const v8bf*)p;
    v8bf hi = *(const v8bf*)(p + 16);
    v16bf f;
#pragma unroll
    for (int i = 0; i < 8; ++i) { f[i] = lo[i]; f[8 + i] = hi[i]; }
    return f;
}

// B-matrix 32x16 bf16 from column-major (K-contiguous) storage.
__device__ __forceinline__ v16bf load_b_frag(const bf16_t* base, int ld, int kb, int lane) {
    int c = lane & 15, h = lane >> 4;
    const bf16_t* p = base + (size_t)c * ld + kb + h * 8;
    v8bf lo = *(const v8bf*)p;
    v8bf hi = *(const v8bf*)(p + 16);
    v16bf f;
#pragma unroll
    for (int i = 0; i < 8; ++i) { f[i] = lo[i]; f[8 + i] = hi[i]; }
    return f;
}

// Async global->LDS copy of one 16-byte chunk per lane (ASYNCcnt-tracked DMA).
__device__ __forceinline__ void async_g2l_b128(const bf16_t* gsrc, bf16_t* ldst) {
    unsigned lds_off = (unsigned)(size_t)ldst;                   // LDS addr = low 32 bits
    unsigned long long ga = (unsigned long long)(size_t)gsrc;
    asm volatile("global_load_async_to_lds_b128 %0, %1, off"
                 :: "v"(lds_off), "v"(ga) : "memory");
}
__device__ __forceinline__ void wait_asynccnt0() {
    asm volatile("s_wait_asynccnt 0x0" ::: "memory");
}

// ---------------- kernel 0a: fp32 -> bf16 convert ----------------
__global__ void k_convert(const float* __restrict__ src, bf16_t* __restrict__ dst, int n) {
    int i = blockIdx.x * blockDim.x + threadIdx.x;
    if (i < n) dst[i] = (bf16_t)src[i];
}

// ---------------- kernel 0b: W[k][c] fp32 -> Wt[c][k] bf16 ----------------
__global__ void k_transpose(const float* __restrict__ W, bf16_t* __restrict__ Wt, int K, int C) {
    int idx = blockIdx.x * blockDim.x + threadIdx.x;
    if (idx < K * C) {
        int c = idx / K, k = idx - c * K;
        Wt[idx] = (bf16_t)W[(size_t)k * C + c];
    }
}

// ---------------- kernel 1: projection GEMM (software-pipelined) ----------------
__global__ __launch_bounds__(128, 1)
void k_proj(const bf16_t* __restrict__ A, const bf16_t* __restrict__ Wt,
            bf16_t* __restrict__ Qh, bf16_t* __restrict__ Vh, bf16_t* __restrict__ Kh,
            int mode) {
    int lane = threadIdx.x & 31, w = threadIdx.x >> 5;
    int rowBase = blockIdx.x * 64 + w * 16;
    int colBase = blockIdx.y * 64;

    v8f acc[4] = {};
    const bf16_t* Arow = A + (size_t)rowBase * DIM;

    v16bf af = load_a_frag(Arow, DIM, 0, lane);
    v16bf bfr[4];
#pragma unroll
    for (int t = 0; t < 4; ++t)
        bfr[t] = load_b_frag(Wt + (size_t)(colBase + t * 16) * DIM, DIM, 0, lane);

#pragma unroll 5
    for (int kb = 0; kb < DIM - 32; kb += 32) {
        v16bf afn = load_a_frag(Arow, DIM, kb + 32, lane);
        v16bf bn[4];
#pragma unroll
        for (int t = 0; t < 4; ++t)
            bn[t] = load_b_frag(Wt + (size_t)(colBase + t * 16) * DIM, DIM, kb + 32, lane);
#pragma unroll
        for (int t = 0; t < 4; ++t) acc[t] = WMMA_BF16(af, bfr[t], acc[t]);
        af = afn;
#pragma unroll
        for (int t = 0; t < 4; ++t) bfr[t] = bn[t];
    }
#pragma unroll
    for (int t = 0; t < 4; ++t) acc[t] = WMMA_BF16(af, bfr[t], acc[t]);

    int c0 = lane & 15, hf = lane >> 4;
#pragma unroll
    for (int t = 0; t < 4; ++t) {
#pragma unroll
        for (int i = 0; i < 8; ++i) {
            int row = rowBase + i + 8 * hf;
            int b = row >> 11, n = row & (Nn - 1);
            int cg = colBase + t * 16 + c0;
            float v = acc[t][i];
            if (mode == 0) {
                if (cg < INNER) {
                    int hh = cg >> 6, d = cg & 63;
                    Qh[(((size_t)(b * H + hh) * Nn) + n) * DH + d] = (bf16_t)v;
                } else {
                    int c2 = cg - INNER, hh = c2 >> 6, d = c2 & 63;
                    Vh[(((size_t)(b * H + hh) * Nn) + n) * DH + d] = (bf16_t)v;
                }
            } else {
                int hh = cg >> 6, d = cg & 63;
                Kh[(((size_t)(b * H + hh) * Mm) + n) * DH + d] = (bf16_t)v;
            }
        }
    }
}

// ---------------- kernel 2: fused flash attention ----------------
// grid.x = B*H (16), grid.y = N/128 (16); 256 threads = 8 waves, 16 rows/wave.
// Double-buffered LDS tiles: K staged by async-to-LDS DMA, V transposed via regs,
// next tile's traffic overlapped with current tile's WMMA + softmax.
__global__ __launch_bounds__(256, 1)
void k_attn(const bf16_t* __restrict__ Qh, const bf16_t* __restrict__ Kh,
            const bf16_t* __restrict__ Vh, const float* __restrict__ attn_mat,
            const float* __restrict__ dots_para, const float* __restrict__ mat_para,
            bf16_t* __restrict__ O) {
    __shared__ __align__(16) bf16_t Klds[2][64 * 72];
    __shared__ __align__(16) bf16_t VTlds[2][64 * 72];
    __shared__ __align__(16) bf16_t Plds[8][16 * 72];

    int tid = threadIdx.x, lane = tid & 31, w = tid >> 5;
    int bh = blockIdx.x;
    int b = bh >> 3, h = bh & 7;
    int nBase = blockIdx.y * 128 + w * 16;
    int c0 = lane & 15, hf = lane >> 4;

    float dscale = 0.125f * dots_para[0];  // DH^-0.5 * dots_para
    float mpar = mat_para[0];

    const bf16_t* Qbase = Qh + ((size_t)bh * Nn + nBase) * DH;
    v16bf qf0 = load_a_frag(Qbase, DH, 0, lane);
    v16bf qf1 = load_a_frag(Qbase, DH, 32, lane);

    const bf16_t* Kb = Kh + (size_t)bh * Mm * DH;
    const bf16_t* Vb = Vh + (size_t)bh * Mm * DH;
    const float* Ab = attn_mat + ((size_t)bh * Nn + nBase) * Mm;

    // staging coordinates: 512 chunks of 16B for K; 64x64 V tile, 32B per thread
    int kml0 = tid >> 3,         kdp0 = (tid & 7) * 8;          // chunk tid
    int kml1 = (tid + 256) >> 3, kdp1 = ((tid + 256) & 7) * 8;  // chunk tid+256
    int smloc = tid >> 2, sdp = (tid & 3) * 16;

    v8f oacc[4] = {};
    float mrow[8], lrow[8];
#pragma unroll
    for (int i = 0; i < 8; ++i) { mrow[i] = -1e30f; lrow[i] = 0.f; }

    // prologue: stage tile mb=0 into buffer 0
    async_g2l_b128(Kb + (size_t)kml0 * DH + kdp0, &Klds[0][kml0 * 72 + kdp0]);
    async_g2l_b128(Kb + (size_t)kml1 * DH + kdp1, &Klds[0][kml1 * 72 + kdp1]);
    {
        const bf16_t* vs = Vb + (size_t)smloc * DH + sdp;
        v8bf va = *(const v8bf*)vs, vb2 = *(const v8bf*)(vs + 8);
#pragma unroll
        for (int j = 0; j < 8; ++j) {
            VTlds[0][(sdp + j) * 72 + smloc]     = va[j];
            VTlds[0][(sdp + 8 + j) * 72 + smloc] = vb2[j];
        }
    }
    wait_asynccnt0();
    __syncthreads();

    for (int mb = 0; mb < Mm; mb += 64) {
        int cur = (mb >> 6) & 1;
        bool pref = (mb + 64) < Mm;
        v8bf nva = {}, nvb = {};
        if (pref) {
            // issue next tile: async K DMA + V global loads (consumed later)
            const bf16_t* Kn = Kb + (size_t)(mb + 64) * DH;
            async_g2l_b128(Kn + (size_t)kml0 * DH + kdp0, &Klds[cur ^ 1][kml0 * 72 + kdp0]);
            async_g2l_b128(Kn + (size_t)kml1 * DH + kdp1, &Klds[cur ^ 1][kml1 * 72 + kdp1]);
            const bf16_t* vs = Vb + (size_t)(mb + 64 + smloc) * DH + sdp;
            nva = *(const v8bf*)vs;
            nvb = *(const v8bf*)(vs + 8);
        }

        // ---- dots = Q @ K^T (K=64 in two 32-chunks) + bias ----
        v8f s[4];
#pragma unroll
        for (int t = 0; t < 4; ++t) {
            v8f acc = {};
            v16bf bk0 = load_b_frag(&Klds[cur][(t * 16) * 72], 72, 0, lane);
            v16bf bk1 = load_b_frag(&Klds[cur][(t * 16) * 72], 72, 32, lane);
            acc = WMMA_BF16(qf0, bk0, acc);
            acc = WMMA_BF16(qf1, bk1, acc);
#pragma unroll
            for (int i = 0; i < 8; ++i) {
                int row = i + 8 * hf;
                float am = __builtin_nontemporal_load(Ab + (size_t)row * Mm + mb + t * 16 + c0);
                acc[i] = acc[i] * dscale + am * mpar;
            }
            s[t] = acc;
        }

        // ---- online softmax (rows span the 16 lanes of each wave half) ----
        float resc[8];
#pragma unroll
        for (int i = 0; i < 8; ++i) {
            float rm = fmaxf(fmaxf(s[0][i], s[1][i]), fmaxf(s[2][i], s[3][i]));
#pragma unroll
            for (int msk = 1; msk < 16; msk <<= 1) rm = fmaxf(rm, __shfl_xor(rm, msk, 32));
            float mnew = fmaxf(mrow[i], rm);
            float r = __expf(mrow[i] - mnew);
            mrow[i] = mnew;
            resc[i] = r;
            float rs = 0.f;
#pragma unroll
            for (int t = 0; t < 4; ++t) {
                float p = __expf(s[t][i] - mnew);
                s[t][i] = p;
                rs += p;
            }
#pragma unroll
            for (int msk = 1; msk < 16; msk <<= 1) rs += __shfl_xor(rs, msk, 32);
            lrow[i] = lrow[i] * r + rs;
        }
#pragma unroll
        for (int t = 0; t < 4; ++t)
#pragma unroll
            for (int i = 0; i < 8; ++i) oacc[t][i] *= resc[i];

        // ---- probs: C-layout -> LDS -> A-layout (wave-private region) ----
        bf16_t* Pw = &Plds[w][0];
#pragma unroll
        for (int t = 0; t < 4; ++t)
#pragma unroll
            for (int i = 0; i < 8; ++i)
                Pw[(i + 8 * hf) * 72 + t * 16 + c0] = (bf16_t)s[t][i];

        // ---- out += P @ V ----
#pragma unroll
        for (int mc = 0; mc < 2; ++mc) {
            v16bf pf = load_a_frag(Pw, 72, mc * 32, lane);
            v16bf vf[4];
#pragma unroll
            for (int dt = 0; dt < 4; ++dt)
                vf[dt] = load_b_frag(&VTlds[cur][(dt * 16) * 72], 72, mc * 32, lane);
#pragma unroll
            for (int dt = 0; dt < 4; ++dt)
                oacc[dt] = WMMA_BF16(pf, vf[dt], oacc[dt]);
        }

        // ---- commit next V tile (transposed ds stores), drain DMA, barrier ----
        if (pref) {
#pragma unroll
            for (int j = 0; j < 8; ++j) {
                VTlds[cur ^ 1][(sdp + j) * 72 + smloc]     = nva[j];
                VTlds[cur ^ 1][(sdp + 8 + j) * 72 + smloc] = nvb[j];
            }
        }
        wait_asynccnt0();
        __syncthreads();
    }

    // ---- normalize and store O[b][n][h*64+d] bf16 ----
    bf16_t* Ob = O + ((size_t)b * Nn + nBase) * INNER + h * DH;
#pragma unroll
    for (int i = 0; i < 8; ++i) lrow[i] = 1.0f / lrow[i];
#pragma unroll
    for (int dt = 0; dt < 4; ++dt)
#pragma unroll
        for (int i = 0; i < 8; ++i) {
            int row = i + 8 * hf;
            Ob[(size_t)row * INNER + dt * 16 + c0] = (bf16_t)(oacc[dt][i] * lrow[i]);
        }
}

// ---------------- kernel 3: output projection + bias (software-pipelined) ----------------
__global__ __launch_bounds__(256, 1)
void k_out(const bf16_t* __restrict__ O, const bf16_t* __restrict__ Wt,
           const float* __restrict__ bias, float* __restrict__ out) {
    int lane = threadIdx.x & 31, w = threadIdx.x >> 5;
    int rowBase = blockIdx.x * 128 + w * 16;
    int colBase = blockIdx.y * 64;

    v8f acc[4] = {};
    const bf16_t* Arow = O + (size_t)rowBase * INNER;

    v16bf af = load_a_frag(Arow, INNER, 0, lane);
    v16bf bfr[4];
#pragma unroll
    for (int t = 0; t < 4; ++t)
        bfr[t] = load_b_frag(Wt + (size_t)(colBase + t * 16) * INNER, INNER, 0, lane);

#pragma unroll 5
    for (int kb = 0; kb < INNER - 32; kb += 32) {
        v16bf afn = load_a_frag(Arow, INNER, kb + 32, lane);
        v16bf bn[4];
#pragma unroll
        for (int t = 0; t < 4; ++t)
            bn[t] = load_b_frag(Wt + (size_t)(colBase + t * 16) * INNER, INNER, kb + 32, lane);
#pragma unroll
        for (int t = 0; t < 4; ++t) acc[t] = WMMA_BF16(af, bfr[t], acc[t]);
        af = afn;
#pragma unroll
        for (int t = 0; t < 4; ++t) bfr[t] = bn[t];
    }
#pragma unroll
    for (int t = 0; t < 4; ++t) acc[t] = WMMA_BF16(af, bfr[t], acc[t]);

    int c0 = lane & 15, hf = lane >> 4;
#pragma unroll
    for (int t = 0; t < 4; ++t) {
        float bv = bias[colBase + t * 16 + c0];
#pragma unroll
        for (int i = 0; i < 8; ++i) {
            int row = rowBase + i + 8 * hf;
            out[(size_t)row * DIM + colBase + t * 16 + c0] = acc[t][i] + bv;
        }
    }
}

// ---------------- host launch ----------------
extern "C" void kernel_launch(void* const* d_in, const int* in_sizes, int n_in,
                              void* d_out, int out_size, void* d_ws, size_t ws_size,
                              hipStream_t stream) {
    const float* x    = (const float*)d_in[0];
    const float* x1   = (const float*)d_in[1];
    const float* amat = (const float*)d_in[2];
    const float* dpar = (const float*)d_in[3];
    const float* mpar = (const float*)d_in[4];
    const float* Wqv  = (const float*)d_in[5];
    const float* Wk   = (const float*)d_in[6];
    const float* Wout = (const float*)d_in[7];
    const float* bout = (const float*)d_in[8];
    float* out = (float*)d_out;

    char* ws = (char*)d_ws;
    bf16_t* x_bf   = (bf16_t*)ws;                 ws += (size_t)Bb * Nn * DIM * 2;      // 4 MB
    bf16_t* x1_bf  = (bf16_t*)ws;                 ws += (size_t)Bb * Mm * DIM * 2;      // 4 MB
    bf16_t* Wqv_t  = (bf16_t*)ws;                 ws += (size_t)DIM * 2 * INNER * 2;    // 1 MB
    bf16_t* Wk_t   = (bf16_t*)ws;                 ws += (size_t)DIM * INNER * 2;        // 512 KB
    bf16_t* Wout_t = (bf16_t*)ws;                 ws += (size_t)INNER * DIM * 2;        // 512 KB
    bf16_t* Qh     = (bf16_t*)ws;                 ws += (size_t)Bb * H * Nn * DH * 2;   // 4 MB
    bf16_t* Kh     = (bf16_t*)ws;                 ws += (size_t)Bb * H * Mm * DH * 2;   // 4 MB
    bf16_t* Vh     = (bf16_t*)ws;                 ws += (size_t)Bb * H * Mm * DH * 2;   // 4 MB
    bf16_t* Obuf   = (bf16_t*)ws;                 // 4 MB

    int nx = Bb * Nn * DIM;
    k_convert<<<(nx + 255) / 256, 256, 0, stream>>>(x, x_bf, nx);
    k_convert<<<(nx + 255) / 256, 256, 0, stream>>>(x1, x1_bf, nx);

    int nqv = DIM * 2 * INNER;
    k_transpose<<<(nqv + 255) / 256, 256, 0, stream>>>(Wqv, Wqv_t, DIM, 2 * INNER);
    int nk = DIM * INNER;
    k_transpose<<<(nk + 255) / 256, 256, 0, stream>>>(Wk, Wk_t, DIM, INNER);
    k_transpose<<<(nk + 255) / 256, 256, 0, stream>>>(Wout, Wout_t, INNER, DIM);

    // qv projection: 4096 rows x 1024 cols
    k_proj<<<dim3(Bb * Nn / 64, 2 * INNER / 64), 128, 0, stream>>>(x_bf, Wqv_t, Qh, Vh, Kh, 0);
    // k projection: 4096 rows x 512 cols
    k_proj<<<dim3(Bb * Mm / 64, INNER / 64), 128, 0, stream>>>(x1_bf, Wk_t, Qh, Vh, Kh, 1);

    // fused attention
    k_attn<<<dim3(Bb * H, Nn / 128), 256, 0, stream>>>(Qh, Kh, Vh, amat, dpar, mpar, Obuf);

    // output projection
    k_out<<<dim3(Bb * Nn / 128, DIM / 64), 256, 0, stream>>>(Obuf, Wout_t, bout, out);
}
